// myModel_12360915878228
// MI455X (gfx1250) — matrix operand order repaired and checked by reading, not compiled, not run on hardware
//
#include <hip/hip_runtime.h>

// ---------------------------------------------------------------------------
// Types for CDNA5 WMMA (wave32): v_wmma_f32_16x16x32_bf16
// ---------------------------------------------------------------------------
typedef __bf16 bf16_t;
typedef bf16_t v16bf __attribute__((ext_vector_type(16)));
typedef float  v8f   __attribute__((ext_vector_type(8)));

__device__ __forceinline__ unsigned int f2bf(float f) {
    union { float f; unsigned int u; } x; x.f = f;
    unsigned int r = x.u + 0x7FFFu + ((x.u >> 16) & 1u);   // round-to-nearest-even
    return r >> 16;
}
// pack two floats into 2xbf16; prefer HW v_cvt_pk_bf16_f32 when available
__device__ __forceinline__ unsigned int pack2bf(float lo, float hi) {
#if __has_builtin(__builtin_amdgcn_cvt_pk_bf16_f32)
    typedef __bf16 v2bf __attribute__((ext_vector_type(2)));
    union { v2bf v; unsigned int u; } r;
    r.v = __builtin_amdgcn_cvt_pk_bf16_f32(lo, hi);
    return r.u;
#else
    return f2bf(lo) | (f2bf(hi) << 16);
#endif
}
// fast sigmoid: v_exp_f32 + v_add + v_rcp_f32 (avoid IEEE divide expansion)
__device__ __forceinline__ float sigmoidf_(float x) {
    return __builtin_amdgcn_rcpf(1.0f + __expf(-x));
}

// ---------------------------------------------------------------------------
// Zero-fill (accumulators must be 0 before atomic scatter-add)
// ---------------------------------------------------------------------------
__global__ __launch_bounds__(256) void zero_f32_kernel(float* __restrict__ p, long n4) {
    long i = (long)blockIdx.x * blockDim.x + threadIdx.x;
    if (i < n4) {
        float4 z = {0.f, 0.f, 0.f, 0.f};
        *(float4*)(p + i * 4) = z;
    }
}

// ---------------------------------------------------------------------------
// COO SpMM scatter:  Y[b][dst[e]][:] += vals[e] * X[src[e]][:]   (D = 128)
// One wave32 per edge; lane owns 4 consecutive floats. Embedding tables are
// small (<=51 MB) and hot in the 192 MB L2; scatters use HW f32 atomics.
// ---------------------------------------------------------------------------
__global__ __launch_bounds__(256) void spmm_kernel(
    const int*   __restrict__ dst,   // [B*NNZ]
    const int*   __restrict__ src,   // [B*NNZ]
    const float* __restrict__ vals,  // [B*NNZ]
    const float* __restrict__ X,     // [n_src, 128]
    float*       __restrict__ Y,     // [B, n_dst, 128]
    long nnz_per_b, long y_bstride, long n_edges)
{
    long wid = ((long)blockIdx.x * blockDim.x + threadIdx.x) >> 5;
    int lane = threadIdx.x & 31;
    if (wid >= n_edges) return;

    long b = wid / nnz_per_b;
    int  d = dst[wid];
    int  s = src[wid];
    float v = vals[wid];

    float4 x = *(const float4*)(X + (long)s * 128 + lane * 4);
    float* y = Y + b * y_bstride + (long)d * 128 + lane * 4;
    unsafeAtomicAdd(y + 0, v * x.x);
    unsafeAtomicAdd(y + 1, v * x.y);
    unsafeAtomicAdd(y + 2, v * x.z);
    unsafeAtomicAdd(y + 3, v * x.w);
}

// ---------------------------------------------------------------------------
// Fused GEMM:  Y[row, 0:128] = act( A[row, 0:K] @ W[0:K, 0:128] )
//
//  AMODE 0 : A is f32, plain rows (stride astride)
//  AMODE 1 : A is f32, row = 0.25 * sum_{b<4} rows at b*bstride  (mean over B)
//  AMODE 2 : A is bf16 bits (ushort), plain rows
//  SIG     : apply sigmoid
//  OF32    : write f32 output (row stride 128)
//  OBF     : write bf16 output (row stride obf_stride; col offset folded into
//            the pointer -> realizes the layer concat for free)
//
// Block = 256 threads = 8 waves; computes a 16-row x 128-col tile.
// LDS tiles are row-padded by 8 ushorts (16 B) so ds_load_b128 fragment
// reads map row r to banks 4r+c (mod 64): A-frag conflict-free, B-frag
// read with b16 loads.
//
//  A frag (lane=M, K packed in VGPRs):  2x b128 from Atile
//  B frag (lane=N, K strided by WROW):  16x b16 from Wlds kept [K][N]
// ---------------------------------------------------------------------------
template<int AMODE, bool SIG, int K, bool OF32, bool OBF>
__global__ __launch_bounds__(256) void gemm_wmma_kernel(
    const void*  __restrict__ Asrc, long astride, long bstride,
    const float* __restrict__ W,                    // [K,128] f32
    unsigned short* __restrict__ out_bf, long obf_stride,
    float* __restrict__ out_f32,                    // row stride 128
    int nrows)
{
    constexpr int WROW = 128 + 8;   // padded W row, ushorts
    constexpr int AROW = K + 8;     // padded A row, ushorts

    extern __shared__ char smem[];
    unsigned short* Wlds  = (unsigned short*)smem;                 // K rows x WROW
    unsigned short* Atile = (unsigned short*)(smem + (size_t)K * WROW * 2);

    const int tid  = threadIdx.x;
    const int row0 = blockIdx.x * 16;

    // --- stage W (f32 -> packed 2xbf16 dword writes) -----------------------
    {
        unsigned int* Wd = (unsigned int*)Wlds;
        for (int i = tid; i < K * 64; i += 256) {
            int k = i >> 6, np = i & 63;
            float2 w = *(const float2*)(W + (long)k * 128 + np * 2);
            Wd[k * (WROW / 2) + np] = pack2bf(w.x, w.y);
        }
    }
    // --- stage 16 x K A tile (with optional mean-over-4 fusion) ------------
    {
        unsigned int* Ad = (unsigned int*)Atile;
        for (int i = tid; i < 16 * (K / 2); i += 256) {
            int m = i / (K / 2), kp = i - m * (K / 2), k = kp * 2;
            int r = row0 + m; if (r >= nrows) r = nrows - 1;
            unsigned int bits;
            if (AMODE == 0) {
                float2 a = *(const float2*)((const float*)Asrc + (long)r * astride + k);
                bits = pack2bf(a.x, a.y);
            } else if (AMODE == 1) {
                const float* A = (const float*)Asrc;
                long base = (long)r * astride + k;
                float2 a0 = *(const float2*)(A + base);
                float2 a1 = *(const float2*)(A + base + bstride);
                float2 a2 = *(const float2*)(A + base + 2 * bstride);
                float2 a3 = *(const float2*)(A + base + 3 * bstride);
                bits = pack2bf(0.25f * (a0.x + a1.x + a2.x + a3.x),
                               0.25f * (a0.y + a1.y + a2.y + a3.y));
            } else {
                bits = *(const unsigned int*)((const unsigned short*)Asrc +
                                              (long)r * astride + k);
            }
            Ad[m * (AROW / 2) + kp] = bits;
        }
    }
    __syncthreads();

    const int wave = tid >> 5;     // 0..7 -> N tile base = 16*wave
    const int lane = tid & 31;

    // ISA fragment coordinates (wave32, 16x16x32 bf16):
    //  A: lane = M (l&15); half h = l>>4; elems 0-7 -> K = 32kb + 8h + e,
    //     elems 8-15 -> K = 32kb + 16 + 8h + (e-8)          (2x b128 loads)
    //  B: lane = N (l&15); elems same K order as A, gathered from
    //     Wlds[K][N] with stride WROW                        (16x b16 loads)
    const int mA = lane & 15;
    const int hA = lane >> 4;
    const int kB = (lane & 15) + 16 * (lane >> 4);

    v8f acc = {};
    #pragma unroll
    for (int kb = 0; kb < K / 32; ++kb) {
        union { v16bf v; int4 q[2]; unsigned short u[16]; } a, b;
        const char* ab = (const char*)(Atile + mA * AROW + kb * 32 + hA * 8);
        a.q[0] = *(const int4*)(ab);
        a.q[1] = *(const int4*)(ab + 32);
        const unsigned short* bp = Wlds + (kb * 32 + hA * 8) * WROW + wave * 16 + mA;
        #pragma unroll
        for (int s = 0; s < 8; ++s) {
            b.u[s]     = bp[s * WROW];
            b.u[8 + s] = bp[(16 + s) * WROW];
        }
        acc = __builtin_amdgcn_wmma_f32_16x16x32_bf16(
                  false, a.v, false, b.v, (short)0, acc, false, false);
    }

    // --- epilogue: sigmoid + stores ----------------------------------------
    // C/D layout: lane l -> n = l&15 ; elem j -> m = j + 8*(l>>4)
    const int n  = (lane & 15) + wave * 16;
    const int mh = (lane >> 4) * 8;

    if (row0 + 16 <= nrows) {               // full tile: straight-line stores
        #pragma unroll
        for (int j = 0; j < 8; ++j) {
            int m = row0 + mh + j;
            float y = acc[j];
            if (SIG) y = sigmoidf_(y);
            if (OF32) out_f32[(long)m * 128 + n] = y;
            if (OBF)  out_bf [(long)m * obf_stride + n] = (unsigned short)pack2bf(y, 0.f);
        }
    } else {                                 // tail tile (defensive)
        #pragma unroll
        for (int j = 0; j < 8; ++j) {
            int m = row0 + mh + j;
            if (m >= nrows) continue;
            float y = acc[j];
            if (SIG) y = sigmoidf_(y);
            if (OF32) out_f32[(long)m * 128 + n] = y;
            if (OBF)  out_bf [(long)m * obf_stride + n] = (unsigned short)pack2bf(y, 0.f);
        }
    }
}

// ---------------------------------------------------------------------------
// Host side
// ---------------------------------------------------------------------------
template<int AMODE, bool SIG, int K, bool OF32, bool OBF>
static void launch_gemm(const void* A, long astride, long bstride, const float* W,
                        unsigned short* obf, long obf_stride, float* of32,
                        int nrows, hipStream_t s)
{
    dim3 grid((nrows + 15) / 16), block(256);
    size_t smem = (size_t)K * (128 + 8) * 2 + (size_t)16 * (K + 8) * 2;
    hipLaunchKernelGGL((gemm_wmma_kernel<AMODE, SIG, K, OF32, OBF>),
                       grid, block, smem, s,
                       A, astride, bstride, W, obf, obf_stride, of32, nrows);
}

static void launch_zero(float* p, long n, hipStream_t s) {
    long n4 = n / 4;
    dim3 grid((unsigned)((n4 + 255) / 256)), block(256);
    hipLaunchKernelGGL(zero_f32_kernel, grid, block, 0, s, p, n4);
}

static void launch_spmm(const int* dst, const int* src, const float* vals,
                        const float* X, float* Y, long nnz_per_b, long y_bstride,
                        long n_edges, hipStream_t s)
{
    dim3 grid((unsigned)((n_edges * 32 + 255) / 256)), block(256);
    hipLaunchKernelGGL(spmm_kernel, grid, block, 0, s,
                       dst, src, vals, X, Y, nnz_per_b, y_bstride, n_edges);
}

extern "C" void kernel_launch(void* const* d_in, const int* in_sizes, int n_in,
                              void* d_out, int out_size, void* d_ws, size_t ws_size,
                              hipStream_t stream)
{
    const int Dm = 128, Bm = 4;
    const int U = in_sizes[0] / Dm;           // 100000
    const int I = in_sizes[1] / Dm;           // 50000
    const long NNZt = in_sizes[2];            // B * NNZ
    const long NNZ  = NNZt / Bm;

    const float* user_emb = (const float*)d_in[0];
    const float* item_emb = (const float*)d_in[1];
    const int*   rows     = (const int*)  d_in[2];
    const int*   cols     = (const int*)  d_in[3];
    const float* vals     = (const float*)d_in[4];
    const float* u_w      = (const float*)d_in[5];   // [2,128,128]
    const float* i_w      = (const float*)d_in[6];
    const float* u_cat_w  = (const float*)d_in[7];   // [256,128]
    const float* i_cat_w  = (const float*)d_in[8];

    // ---- workspace layout -------------------------------------------------
    char* p = (char*)d_ws;
    float* ues = (float*)p;           p += (size_t)Bm * U * Dm * 4;   // f32 accum
    float* ies = (float*)p;           p += (size_t)Bm * I * Dm * 4;
    float* ue_cur = (float*)p;        p += (size_t)U * Dm * 4;        // layer carry
    float* ie_cur = (float*)p;        p += (size_t)I * Dm * 4;
    unsigned short* mu_cat = (unsigned short*)p; p += (size_t)U * 2 * Dm * 2;      // bf16 concat
    unsigned short* mi_cat = (unsigned short*)p; p += (size_t)I * 2 * Dm * 2;
    unsigned short* su_cat = (unsigned short*)p; p += (size_t)Bm * U * 2 * Dm * 2;
    unsigned short* si_cat = (unsigned short*)p; p += (size_t)Bm * I * 2 * Dm * 2;

    const long ubstr = (long)U * Dm;  // behavior stride in ues
    const long ibstr = (long)I * Dm;

    const float* uX = item_emb;   // layer-0 SpMM gathers
    const float* iX = user_emb;

    for (int l = 0; l < 2; ++l) {
        const float* uw = u_w + (size_t)l * Dm * Dm;
        const float* iw = i_w + (size_t)l * Dm * Dm;

        // ues[b][r] += v * X_items[c] ; ies[b][c] += v * X_users[r]
        launch_zero(ues, (long)Bm * U * Dm, stream);
        launch_zero(ies, (long)Bm * I * Dm, stream);
        launch_spmm(rows, cols, vals, uX, ues, NNZ, ubstr, NNZt, stream);
        launch_spmm(cols, rows, vals, iX, ies, NNZ, ibstr, NNZt, stream);

        // su/si : per-behavior sigmoid GEMM, written as concat columns l*128
        launch_gemm<0, true, 128, false, true>(
            ues, Dm, 0, uw, su_cat + l * Dm, 2 * Dm, nullptr, Bm * U, stream);
        launch_gemm<0, true, 128, false, true>(
            ies, Dm, 0, iw, si_cat + l * Dm, 2 * Dm, nullptr, Bm * I, stream);

        // ue/ie : mean-over-behaviors fused into A load; f32 carry only needed
        // after layer 0 (it feeds layer-1 SpMM gathers)
        if (l == 0) {
            launch_gemm<1, true, 128, true, true>(
                ues, Dm, ubstr, uw, mu_cat + l * Dm, 2 * Dm, ue_cur, U, stream);
            launch_gemm<1, true, 128, true, true>(
                ies, Dm, ibstr, iw, mi_cat + l * Dm, 2 * Dm, ie_cur, I, stream);
        } else {
            launch_gemm<1, true, 128, false, true>(
                ues, Dm, ubstr, uw, mu_cat + l * Dm, 2 * Dm, nullptr, U, stream);
            launch_gemm<1, true, 128, false, true>(
                ies, Dm, ibstr, iw, mi_cat + l * Dm, 2 * Dm, nullptr, I, stream);
        }

        uX = ie_cur;   // layer-1 SpMM gathers from updated embeddings
        iX = ue_cur;
    }

    // ---- final concat GEMMs straight into d_out ---------------------------
    float* out = (float*)d_out;
    float* user_embed  = out;
    float* item_embed  = out + (size_t)U * Dm;
    float* user_embeds = out + (size_t)(U + I) * Dm;
    float* item_embeds = out + (size_t)(U + I + (size_t)Bm * U) * Dm;

    launch_gemm<2, false, 256, true, false>(
        mu_cat, 2 * Dm, 0, u_cat_w, nullptr, 0, user_embed, U, stream);
    launch_gemm<2, false, 256, true, false>(
        mi_cat, 2 * Dm, 0, i_cat_w, nullptr, 0, item_embed, I, stream);
    launch_gemm<2, false, 256, true, false>(
        su_cat, 2 * Dm, 0, u_cat_w, nullptr, 0, user_embeds, Bm * U, stream);
    launch_gemm<2, false, 256, true, false>(
        si_cat, 2 * Dm, 0, i_cat_w, nullptr, 0, item_embeds, Bm * I, stream);

    (void)n_in; (void)out_size; (void)ws_size;
}
